// GRUDecoder_4518305596072
// MI455X (gfx1250) — matrix operand order repaired and checked by reading, not compiled, not run on hardware
//
#include <hip/hip_runtime.h>

typedef _Float16 f16;
typedef __attribute__((ext_vector_type(16))) _Float16 v16h;
typedef __attribute__((ext_vector_type(8)))  _Float16 v8h;
typedef __attribute__((ext_vector_type(8)))  float    v8f;

#define B_    16
#define T_    25
#define N_    100
#define K_    2
#define MSG_  64
#define GRU_  64
#define NHID_ 128
#define E_    (N_*(N_-1))   // 9900
#define EPN   (N_-1)        // 99 edges per receiver
#define ROWS  112           // 99 padded to 7 tiles of 16
#define MT    7

__device__ __forceinline__ v16h cat8(v8h lo, v8h hi) {
  return __builtin_shufflevector(lo, hi, 0,1,2,3,4,5,6,7,8,9,10,11,12,13,14,15);
}
__device__ __forceinline__ v8f wmma_f16(v16h a, v16h b, v8f c) {
  // D = A(16x32,f16) * B(32x16,f16) + C(16x16,f32)
  return __builtin_amdgcn_wmma_f32_16x16x32_f16(false, a, false, b, (short)0, c, false, false);
}
// Branchless fast tanh/sigmoid: native v_exp_f32 + v_rcp_f32, no exec divergence.
__device__ __forceinline__ float fast_tanh(float x) {
  float xc = fminf(fmaxf(x, -9.f), 9.f);   // keep exp finite -> no inf/inf NaN
  float e  = __expf(2.f * xc);
  return (e - 1.f) * __builtin_amdgcn_rcpf(e + 1.f);
}
__device__ __forceinline__ float sigm(float x) {
  return __builtin_amdgcn_rcpf(1.f + __expf(-x));
}

// ---------------- setup kernels ----------------

__global__ void k_zero(float* h, f16* h16, int nF, int nH) {
  int i = blockIdx.x * blockDim.x + threadIdx.x;
  if (i < nF) h[i]  = 0.f;
  if (i < nH) h16[i] = (f16)0.f;
}

// Pack row-major f32 W[Kdim][Ndim] into WMMA B-fragment order (f16):
// dst[((nt*numK+ks)*32 + lane)*16 + h] = W[ks*32 + (lane>=16?8:0) + (h&7) + (h>=8?16:0)][nt*16 + lane%16]
__global__ void k_pack(const float* __restrict__ src, f16* __restrict__ dst,
                       int Kdim, int Ndim) {
  int idx = blockIdx.x * blockDim.x + threadIdx.x;
  int total = Kdim * Ndim;
  if (idx >= total) return;
  int h    = idx & 15;
  int lane = (idx >> 4) & 31;
  int rest = idx >> 9;
  int numK = Kdim >> 5;
  int ks   = rest % numK;
  int nt   = rest / numK;
  int n = nt * 16 + (lane & 15);
  int k = ks * 32 + ((lane & 16) ? 8 : 0) + (h & 7) + ((h & 8) ? 16 : 0);
  dst[idx] = (f16)src[k * Ndim + n];
}

// ---------------- per-step kernels ----------------

// One block (7 waves) per (batch b, receiver i): 99 edges -> messages -> aggregate.
// Wave w owns M-tile w; A-fragments loaded once and reused across all 4 N-tiles.
__global__ void __launch_bounds__(224) k_edge(
    const f16* __restrict__ hidden16, const float* __restrict__ rel_types,
    const f16* __restrict__ W1p, const f16* __restrict__ W2p,
    const float* __restrict__ b1, const float* __restrict__ b2,
    f16* __restrict__ agg16)
{
  __shared__ __align__(16) f16 m1s[ROWS][MSG_];   // 14 KB
  __shared__ float scaleS[ROWS];
  __shared__ float aggS[MSG_];

  const int blk  = blockIdx.x;
  const int b    = blk / N_;
  const int i    = blk % N_;
  const int tid  = threadIdx.x;
  const int lane = tid & 31;
  const int mt   = tid >> 5;                 // wave id == M-tile (0..6)
  const int lrow = lane & 15;
  const int half16 = (lane & 16) ? 8 : 0;    // A/C layout shift for upper half-wave

  if (tid < ROWS)
    scaleS[tid] = (tid < EPN)
      ? rel_types[((size_t)b * E_ + (size_t)i * EPN + tid) * K_ + 1] * (1.0f / (K_ * N_))
      : 0.0f;   // tmask masks k=0; fold /K and /N into per-edge scale
  if (tid < MSG_) aggS[tid] = 0.f;
  __syncthreads();

  const f16* rowR = hidden16 + (size_t)(b * N_ + i) * GRU_;

  // GEMM1: pre_msg[112x128] x W1[128x64]; A gathered on the fly from hidden16
  {
    const int l = mt * 16 + lrow;
    const int gS = (l < EPN) ? (b * N_ + ((l < i) ? l : l + 1)) : (B_ * N_); // pad row
    const f16* rowS = hidden16 + (size_t)gS * GRU_;
    v16h A[4];
    #pragma unroll
    for (int ks = 0; ks < 4; ++ks) {
      const int c0 = ks * 32 + half16;
      const int c1 = c0 + 16;
      const f16* p0 = (c0 < 64) ? (rowS + c0) : (rowR + (c0 - 64));
      const f16* p1 = (c1 < 64) ? (rowS + c1) : (rowR + (c1 - 64));
      A[ks] = cat8(*(const v8h*)p0, *(const v8h*)p1);
    }
    #pragma unroll
    for (int nt = 0; nt < 4; ++nt) {
      v8f acc = {};
      #pragma unroll
      for (int ks = 0; ks < 4; ++ks)
        acc = wmma_f16(A[ks],
                       *(const v16h*)(W1p + ((size_t)(nt * 4 + ks) * 32 + lane) * 16),
                       acc);
      const int nCol = nt * 16 + lrow;
      const float bias = b1[nCol];
      #pragma unroll
      for (int v = 0; v < 8; ++v)
        m1s[mt * 16 + v + half16][nCol] = (f16)fast_tanh(acc[v] + bias);
    }
  }
  __syncthreads();

  // GEMM2: m1[112x64] x W2[64x64] -> tanh -> *rel_type -> column sums
  {
    v16h A[2];
    #pragma unroll
    for (int ks = 0; ks < 2; ++ks) {
      const int k0 = ks * 32 + half16;
      A[ks] = cat8(*(const v8h*)&m1s[mt * 16 + lrow][k0],
                   *(const v8h*)&m1s[mt * 16 + lrow][k0 + 16]);
    }
    #pragma unroll
    for (int nt = 0; nt < 4; ++nt) {
      v8f acc = {};
      #pragma unroll
      for (int ks = 0; ks < 2; ++ks)
        acc = wmma_f16(A[ks],
                       *(const v16h*)(W2p + ((size_t)(nt * 2 + ks) * 32 + lane) * 16),
                       acc);
      const int nCol = nt * 16 + lrow;
      const float bias = b2[nCol];
      float part = 0.f;
      #pragma unroll
      for (int v = 0; v < 8; ++v)
        part += fast_tanh(acc[v] + bias) * scaleS[mt * 16 + v + half16];
      atomicAdd(&aggS[nCol], part);   // ds_add_f32, block-local edge2node
    }
  }
  __syncthreads();

  if (tid < MSG_) agg16[(size_t)(b * N_ + i) * MSG_ + tid] = (f16)aggS[tid];
}

struct NodeP {
  float* hidden; f16* hidden16; const f16* agg16;
  const float* inputs; float* out; const int* burn;
  const f16 *hrp, *hip_, *hnp, *o1p, *o2p;
  const float *wir, *bir, *wii, *bii, *win, *bin;
  const float *b1o, *b2o, *w3, *b3;
  int t;
};

// One block per 16 nodes: GRU update + output MLP + residual prediction.
__global__ void __launch_bounds__(128) k_node(NodeP P) {
  __shared__ float insS[16];
  __shared__ __align__(16) f16 h16S[16][GRU_];
  __shared__ __align__(16) f16 p1S[16][NHID_];
  __shared__ float p2S[16][NHID_];

  const int tid  = threadIdx.x;
  const int lane = tid & 31;
  const int wave = tid >> 5;            // 4 waves
  const int lrow = lane & 15;
  const int half16 = (lane & 16) ? 8 : 0;
  const int nodeBase = blockIdx.x * 16;

  if (tid < 16) {
    const int g = nodeBase + tid;
    const int bb = g / N_, nn = g % N_;
    const bool useInp = (P.t == 0) || (P.t <= *P.burn);
    insS[tid] = useInp ? P.inputs[((size_t)bb * T_ + P.t) * N_ + nn]
                       : P.out[((size_t)bb * (T_ - 1) + (P.t - 1)) * N_ + nn];
  }
  __syncthreads();

  // GRU gates: 3 GEMMs [16x64]x[64x64]; wave w owns columns [16w,16w+16)
  {
    const int nt = wave;
    v8f aR = {}, aI = {}, aN = {};
    #pragma unroll
    for (int ks = 0; ks < 2; ++ks) {
      const int k0 = ks * 32 + half16;
      const f16* arow = P.agg16 + (size_t)(nodeBase + lrow) * MSG_;
      v16h a = cat8(*(const v8h*)(arow + k0), *(const v8h*)(arow + k0 + 16));
      const size_t fb = ((size_t)(nt * 2 + ks) * 32 + lane) * 16;
      aR = wmma_f16(a, *(const v16h*)(P.hrp + fb), aR);
      aI = wmma_f16(a, *(const v16h*)(P.hip_ + fb), aI);
      aN = wmma_f16(a, *(const v16h*)(P.hnp + fb), aN);
    }
    const int nCol = nt * 16 + lrow;
    const float wir = P.wir[nCol], bir = P.bir[nCol];
    const float wii = P.wii[nCol], bii = P.bii[nCol];
    const float win = P.win[nCol], bin = P.bin[nCol];
    #pragma unroll
    for (int v = 0; v < 8; ++v) {
      const int m = v + half16;
      const int g = nodeBase + m;
      const float ins  = insS[m];
      const float hOld = P.hidden[(size_t)g * GRU_ + nCol];
      const float r  = sigm(ins * wir + bir + aR[v]);
      const float ii = sigm(ins * wii + bii + aI[v]);
      const float nn = fast_tanh(ins * win + bin + r * aN[v]);
      const float hN = (1.f - ii) * nn + ii * hOld;
      P.hidden[(size_t)g * GRU_ + nCol]   = hN;
      P.hidden16[(size_t)g * GRU_ + nCol] = (f16)hN;
      h16S[m][nCol] = (f16)hN;
    }
  }
  __syncthreads();

  // out_fc1: [16x64] x [64x128] + relu (A-frags hoisted out of the nt loop)
  {
    v16h A[2];
    #pragma unroll
    for (int ks = 0; ks < 2; ++ks) {
      const int k0 = ks * 32 + half16;
      A[ks] = cat8(*(const v8h*)&h16S[lrow][k0], *(const v8h*)&h16S[lrow][k0 + 16]);
    }
    for (int nt = wave; nt < 8; nt += 4) {
      v8f acc = {};
      #pragma unroll
      for (int ks = 0; ks < 2; ++ks)
        acc = wmma_f16(A[ks],
                       *(const v16h*)(P.o1p + ((size_t)(nt * 2 + ks) * 32 + lane) * 16),
                       acc);
      const int nCol = nt * 16 + lrow;
      const float bias = P.b1o[nCol];
      #pragma unroll
      for (int v = 0; v < 8; ++v)
        p1S[v + half16][nCol] = (f16)fmaxf(acc[v] + bias, 0.f);
    }
  }
  __syncthreads();

  // out_fc2: [16x128] x [128x128] + relu
  {
    v16h A[4];
    #pragma unroll
    for (int ks = 0; ks < 4; ++ks) {
      const int k0 = ks * 32 + half16;
      A[ks] = cat8(*(const v8h*)&p1S[lrow][k0], *(const v8h*)&p1S[lrow][k0 + 16]);
    }
    for (int nt = wave; nt < 8; nt += 4) {
      v8f acc = {};
      #pragma unroll
      for (int ks = 0; ks < 4; ++ks)
        acc = wmma_f16(A[ks],
                       *(const v16h*)(P.o2p + ((size_t)(nt * 4 + ks) * 32 + lane) * 16),
                       acc);
      const int nCol = nt * 16 + lrow;
      const float bias = P.b2o[nCol];
      #pragma unroll
      for (int v = 0; v < 8; ++v)
        p2S[v + half16][nCol] = fmaxf(acc[v] + bias, 0.f);
    }
  }
  __syncthreads();

  // out_fc3 (128 -> 1) + residual
  if (tid < 16) {
    const int g = nodeBase + tid;
    const int bb = g / N_, nn = g % N_;
    float pred = insS[tid] + P.b3[0];
    #pragma unroll 8
    for (int c = 0; c < NHID_; ++c) pred += p2S[tid][c] * P.w3[c];
    P.out[((size_t)bb * (T_ - 1) + P.t) * N_ + nn] = pred;
  }
}

// ---------------- host ----------------

extern "C" void kernel_launch(void* const* d_in, const int* in_sizes, int n_in,
                              void* d_out, int out_size, void* d_ws, size_t ws_size,
                              hipStream_t stream) {
  (void)in_sizes; (void)n_in; (void)out_size; (void)ws_size;
  const float* inputs    = (const float*)d_in[0];
  const float* rel_types = (const float*)d_in[3];
  const float* msg_fc1_w = (const float*)d_in[4];
  const float* msg_fc1_b = (const float*)d_in[5];
  const float* msg_fc2_w = (const float*)d_in[6];
  const float* msg_fc2_b = (const float*)d_in[7];
  const float* gru_hr_w  = (const float*)d_in[8];
  const float* gru_hi_w  = (const float*)d_in[9];
  const float* gru_hn_w  = (const float*)d_in[10];
  const float* gru_ir_w  = (const float*)d_in[11];
  const float* gru_ir_b  = (const float*)d_in[12];
  const float* gru_ii_w  = (const float*)d_in[13];
  const float* gru_ii_b  = (const float*)d_in[14];
  const float* gru_in_w  = (const float*)d_in[15];
  const float* gru_in_b  = (const float*)d_in[16];
  const float* out_fc1_w = (const float*)d_in[17];
  const float* out_fc1_b = (const float*)d_in[18];
  const float* out_fc2_w = (const float*)d_in[19];
  const float* out_fc2_b = (const float*)d_in[20];
  const float* out_fc3_w = (const float*)d_in[21];
  const float* out_fc3_b = (const float*)d_in[22];
  const int*   burn_steps = (const int*)d_in[24];

  char* ws = (char*)d_ws;
  size_t off = 0;
  auto alloc = [&](size_t bytes) {
    size_t o = off; off = (off + bytes + 255) & ~(size_t)255; return o;
  };
  float* hidden   = (float*)(ws + alloc((size_t)B_ * N_ * GRU_ * 4));
  f16*   hidden16 = (f16*)  (ws + alloc(((size_t)B_ * N_ + 16) * GRU_ * 2)); // +16 zero pad rows
  f16*   agg16    = (f16*)  (ws + alloc((size_t)B_ * N_ * MSG_ * 2));
  f16*   W1p      = (f16*)  (ws + alloc((size_t)128 * 64 * 2));
  f16*   W2p      = (f16*)  (ws + alloc((size_t)64 * 64 * 2));
  f16*   hrp      = (f16*)  (ws + alloc((size_t)64 * 64 * 2));
  f16*   hip_     = (f16*)  (ws + alloc((size_t)64 * 64 * 2));
  f16*   hnp      = (f16*)  (ws + alloc((size_t)64 * 64 * 2));
  f16*   o1p      = (f16*)  (ws + alloc((size_t)64 * 128 * 2));
  f16*   o2p      = (f16*)  (ws + alloc((size_t)128 * 128 * 2));

  const int nF = B_ * N_ * GRU_;
  const int nH = (B_ * N_ + 16) * GRU_;
  k_zero<<<(nH + 255) / 256, 256, 0, stream>>>(hidden, hidden16, nF, nH);

  auto pack = [&](const float* src, f16* dst, int Kd, int Nd) {
    int tot = Kd * Nd;
    k_pack<<<(tot + 255) / 256, 256, 0, stream>>>(src, dst, Kd, Nd);
  };
  pack(msg_fc1_w + 128 * 64, W1p, 128, 64);  // edge-type k=1 slice (k=0 masked out)
  pack(msg_fc2_w + 64 * 64,  W2p, 64, 64);
  pack(gru_hr_w, hrp, 64, 64);
  pack(gru_hi_w, hip_, 64, 64);
  pack(gru_hn_w, hnp, 64, 64);
  pack(out_fc1_w, o1p, 64, 128);
  pack(out_fc2_w, o2p, 128, 128);

  NodeP P;
  P.hidden = hidden; P.hidden16 = hidden16; P.agg16 = agg16;
  P.inputs = inputs; P.out = (float*)d_out; P.burn = burn_steps;
  P.hrp = hrp; P.hip_ = hip_; P.hnp = hnp; P.o1p = o1p; P.o2p = o2p;
  P.wir = gru_ir_w; P.bir = gru_ir_b; P.wii = gru_ii_w; P.bii = gru_ii_b;
  P.win = gru_in_w; P.bin = gru_in_b;
  P.b1o = out_fc1_b; P.b2o = out_fc2_b; P.w3 = out_fc3_w; P.b3 = out_fc3_b;

  const float* b1 = msg_fc1_b + MSG_;  // k=1 slice
  const float* b2 = msg_fc2_b + MSG_;

  for (int t = 0; t < T_ - 1; ++t) {
    k_edge<<<B_ * N_, 224, 0, stream>>>(hidden16, rel_types, W1p, W2p, b1, b2, agg16);
    P.t = t;
    k_node<<<(B_ * N_) / 16, 128, 0, stream>>>(P);
  }
}
